// SingleSelfAttention_29661044146679
// MI455X (gfx1250) — matrix operand order repaired and checked by reading, not compiled, not run on hardware
//
#include <hip/hip_runtime.h>
#include <hip/hip_bf16.h>

// ---------------------------------------------------------------------------
// SingleSelfAttention on gfx1250 (MI455X), bf16 WMMA flash-style.
//
// B=8, T=2048, C=768, NDIM=64.
//   Q = 8 * Z@Wq  (scale sqrt(64) folded into Q)
//   K = Z@Wk ; V = Z@Wv
//   w[q,k] = Q[q].K[k]  (causal: valid iff q >= k)
//   softmax over the QUERY axis (axis=1):  p[q,k] = exp(w-M[k])/D[k]
//   out[q,d] = sum_k p[q,k] V[k,d]
//
// Load balance: causal triangle -> pair tile i with tile 127-i (constant 129
// tiles of work per task) and split each task over the 4 waves of a block,
// merging partials in LDS (online-softmax merge for stats, ds_add_f32 for PV).
//
// Workspace layout (bytes):
//   Qb  bf16 [B*T][64]        @ 0          (2 MiB) row-major
//   Kb  bf16 [B*T][64]        @ 2 MiB      (2 MiB) row-major
//   Vt  bf16 [B][64][T]       @ 4 MiB      (2 MiB) transposed
//   Wt  bf16 [192][768]       @ 6 MiB      (288 KiB) (Wq|Wk|Wv)^T
//   M   f32  [B][T]           @ 6 MiB+288K (64 KiB) column max
//   D   f32  [B][T]           @ +64K       (64 KiB) column expsum
// ---------------------------------------------------------------------------

typedef __attribute__((ext_vector_type(16))) __bf16 v16bf;
typedef __attribute__((ext_vector_type(8)))  __bf16 v8bf;
typedef __attribute__((ext_vector_type(8)))  float  v8f;
typedef __attribute__((ext_vector_type(4)))  float  v4f;

#define T_LEN 2048
#define C_LEN 768
#define NDIM  64
#define NT    (T_LEN / 16)   // 128 tiles

static __device__ __forceinline__ v8bf ld8bf(const __bf16* p) {
    return *(const v8bf*)p;
}
static __device__ __forceinline__ v16bf cat16(v8bf a, v8bf b) {
    return __builtin_shufflevector(a, b, 0,1,2,3,4,5,6,7,8,9,10,11,12,13,14,15);
}
static __device__ __forceinline__ v8f wmma_bf16(v16bf a, v16bf b, v8f c) {
    // D = A(16x32 bf16) * B(32x16 bf16) + C(16x16 f32)
    return __builtin_amdgcn_wmma_f32_16x16x32_bf16(false, a, false, b,
                                                   (short)0, c, false, false);
}

// A-operand loader: 16-bit A 16x32 layout. lane holds row M=lane%16;
// lanes 0-15 get K = [0..7]+[16..23], lanes 16-31 get K = [8..15]+[24..31].
static __device__ __forceinline__ v16bf ldA(const __bf16* row, int k0, int half) {
    return cat16(ld8bf(row + k0 + half * 8),
                 ld8bf(row + k0 + 16 + half * 8));
}
// B-operand loader: 16-bit B 32x16 layout. lane holds col N=lane%16;
// lanes 0-15 get K = [0..7]+[16..23], lanes 16-31 get K = [8..15]+[24..31].
static __device__ __forceinline__ v16bf ldB(const __bf16* col, int k0, int half) {
    return cat16(ld8bf(col + k0 + half * 8),
                 ld8bf(col + k0 + 16 + half * 8));
}

// ---------------------------------------------------------------------------
// Kernel 0: Wt[n][c] = W_{n/64}[c][n%64]  (f32 -> bf16, transpose). 147K elems.
// ---------------------------------------------------------------------------
__global__ void wtrans_kernel(const float* __restrict__ Wq,
                              const float* __restrict__ Wk,
                              const float* __restrict__ Wv,
                              __bf16* __restrict__ Wt) {
    int idx = blockIdx.x * 256 + threadIdx.x;
    if (idx >= 192 * C_LEN) return;
    int n = idx / C_LEN, c = idx % C_LEN;
    int mat = n >> 6, d = n & 63;
    const float* W = (mat == 0) ? Wq : (mat == 1) ? Wk : Wv;
    Wt[idx] = (__bf16)W[c * NDIM + d];
}

// ---------------------------------------------------------------------------
// Kernel 1: QKV projection.  [16384,768] x [768,192] via bf16 WMMA.
// grid (1024, 3), block 128 (4 waves). Each wave: one 16x16 output tile.
// ---------------------------------------------------------------------------
__global__ __launch_bounds__(128) void qkv_kernel(const float* __restrict__ Z,
                                                  const __bf16* __restrict__ Wt,
                                                  __bf16* __restrict__ Qb,
                                                  __bf16* __restrict__ Kb,
                                                  __bf16* __restrict__ Vt) {
    const int lane = threadIdx.x & 31, wave = threadIdx.x >> 5;
    const int col = lane & 15, half = lane >> 4;
    const int mt = blockIdx.x;                 // 0..1023 : 16-token tile
    const int nt = blockIdx.y * 4 + wave;      // 0..11   : 16-col tile of 192
    const int ncol = nt * 16 + col;
    const float*  zrow = Z  + (size_t)(mt * 16 + col) * C_LEN;
    const __bf16* wrow = Wt + (size_t)ncol * C_LEN;

    v8f acc = {};
    for (int c0 = 0; c0 < C_LEN; c0 += 32) {
        // A: convert f32 Z row chunk -> bf16 in the A-operand pattern
        v8f lo = *(const v8f*)(zrow + c0 + half * 8);
        v8f hi = *(const v8f*)(zrow + c0 + 16 + half * 8);
        v16bf a;
#pragma unroll
        for (int i = 0; i < 8; ++i) { a[i] = (__bf16)lo[i]; a[i + 8] = (__bf16)hi[i]; }
        // B: Wt row (already n-major) is the B-operand column
        acc = wmma_bf16(a, ldB(wrow, c0, half), acc);
    }

    const int mat = nt >> 2;     // 0=Q 1=K 2=V (wave-uniform)
    const int d = ncol & 63;
#pragma unroll
    for (int r = 0; r < 8; ++r) {
        const int tok = mt * 16 + r + 8 * half;   // flat b*T + t
        const float v = acc[r];
        if (mat == 0) {
            Qb[(size_t)tok * NDIM + d] = (__bf16)(v * 8.0f);  // fold sqrt(64)
        } else if (mat == 1) {
            Kb[(size_t)tok * NDIM + d] = (__bf16)v;
        } else {
            const int b = tok >> 11, t = tok & (T_LEN - 1);
            Vt[((size_t)(b * NDIM + d)) * T_LEN + t] = (__bf16)v;   // transposed
        }
    }
}

// ---------------------------------------------------------------------------
// Kernel 2: per-key-column stats M[k], D[k] over q in [k, T).
// grid (64, 8), block 128. Block owns key-tile pair (ka, 127-ka): constant 129
// q-tiles of work, strided over the 4 waves; per-wave online (m,d) partials
// merged in LDS with D = sum_w d_w * exp(m_w - M).
// ---------------------------------------------------------------------------
__global__ __launch_bounds__(128) void colstats_kernel(const __bf16* __restrict__ Qb,
                                                       const __bf16* __restrict__ Kb,
                                                       float* __restrict__ Mbuf,
                                                       float* __restrict__ Dbuf) {
    const int lane = threadIdx.x & 31, wave = threadIdx.x >> 5;
    const int col = lane & 15, half = lane >> 4;
    const int ka = blockIdx.x, kb = (NT - 1) - ka;   // paired key tiles
    const int b  = blockIdx.y;
    const size_t base = (size_t)b * T_LEN * NDIM;

    __shared__ float sM[2][4][16];
    __shared__ float sD[2][4][16];

    // B operands: fixed K tiles (column = key token per lane)
    const __bf16* krowA = Kb + base + (size_t)(ka * 16 + col) * NDIM;
    const __bf16* krowB = Kb + base + (size_t)(kb * 16 + col) * NDIM;
    const v16bf bA0 = ldB(krowA, 0, half), bA1 = ldB(krowA, 32, half);
    const v16bf bB0 = ldB(krowB, 0, half), bB1 = ldB(krowB, 32, half);
    const int kcolA = ka * 16 + col, kcolB = kb * 16 + col;
    const int nA = NT - ka, nB = NT - kb;            // q-tiles per member

    float mA = -__builtin_inff(), dA = 0.0f;
    float mB = -__builtin_inff(), dB = 0.0f;

    for (int j = wave; j < nA + nB; j += 4) {        // wave-uniform split
        const bool isA = j < nA;
        const int qt   = isA ? (ka + j) : (kb + (j - nA));
        const int kcol = isA ? kcolA : kcolB;
        const v16bf b0 = isA ? bA0 : bB0;
        const v16bf b1 = isA ? bA1 : bB1;

        const __bf16* qrow = Qb + base + (size_t)(qt * 16 + col) * NDIM;
        v8f s = {};
        s = wmma_bf16(ldA(qrow, 0, half), b0, s);
        s = wmma_bf16(ldA(qrow, 32, half), b1, s);

        float sv[8];
        float tmax = -__builtin_inff();
#pragma unroll
        for (int r = 0; r < 8; ++r) {
            const int q = qt * 16 + r + 8 * half;
            sv[r] = (q >= kcol) ? s[r] : -__builtin_inff();
            tmax = fmaxf(tmax, sv[r]);
        }
        tmax = fmaxf(tmax, __shfl_xor(tmax, 16, 32));   // combine row halves
        const float mOld = isA ? mA : mB;
        const float mnew = fmaxf(mOld, tmax);
        float lsum = 0.0f;
#pragma unroll
        for (int r = 0; r < 8; ++r) lsum += __expf(sv[r] - mnew);
        lsum += __shfl_xor(lsum, 16, 32);
        const float dnew = (isA ? dA : dB) * __expf(mOld - mnew) + lsum;
        if (isA) { mA = mnew; dA = dnew; } else { mB = mnew; dB = dnew; }
    }

    if (lane < 16) {
        sM[0][wave][lane] = mA; sD[0][wave][lane] = dA;
        sM[1][wave][lane] = mB; sD[1][wave][lane] = dB;
    }
    __syncthreads();
    if (threadIdx.x < 32) {
        const int p = threadIdx.x >> 4, c = threadIdx.x & 15;
        float M = fmaxf(fmaxf(sM[p][0][c], sM[p][1][c]),
                        fmaxf(sM[p][2][c], sM[p][3][c]));
        float D = 0.0f;
#pragma unroll
        for (int w = 0; w < 4; ++w) {
            const float dw = sD[p][w][c];
            D += (dw > 0.0f) ? dw * __expf(sM[p][w][c] - M) : 0.0f;
        }
        const int kt = p ? kb : ka;
        Mbuf[b * T_LEN + kt * 16 + c] = M;
        Dbuf[b * T_LEN + kt * 16 + c] = D;
    }
}

// ---------------------------------------------------------------------------
// Kernel 3: output. grid (64, 8), block 128. Block owns query-tile pair
// (qA, 127-qA); its key chunks (32 keys each) are strided over the 4 waves.
// S^T = K*Q^T so its C-layout is exactly the A-operand layout of P for the
// P@V WMMA (no shuffles). Partial PV sums merged via ds_add_f32 in LDS.
// ---------------------------------------------------------------------------
__global__ __launch_bounds__(128) void attn_out_kernel(const __bf16* __restrict__ Qb,
                                                       const __bf16* __restrict__ Kb,
                                                       const __bf16* __restrict__ Vt,
                                                       const float* __restrict__ Mbuf,
                                                       const float* __restrict__ Dbuf,
                                                       float* __restrict__ Out) {
    const int lane = threadIdx.x & 31, wave = threadIdx.x >> 5;
    const int col = lane & 15, half = lane >> 4;
    const int qA = blockIdx.x, qB = (NT - 1) - qA;   // paired query tiles
    const int b  = blockIdx.y;
    const size_t base = (size_t)b * T_LEN * NDIM;

    __shared__ float sOut[2][4][8][32];              // [pair][ntile][r][lane]
    float* sflat = &sOut[0][0][0][0];
    for (int i = threadIdx.x; i < 2048; i += 128) sflat[i] = 0.0f;
    __syncthreads();

    // B operands: Q^T tiles, fixed per block
    const __bf16* qrowA = Qb + base + (size_t)(qA * 16 + col) * NDIM;
    const __bf16* qrowB = Qb + base + (size_t)(qB * 16 + col) * NDIM;
    const v16bf bqA0 = ldB(qrowA, 0, half), bqA1 = ldB(qrowA, 32, half);
    const v16bf bqB0 = ldB(qrowB, 0, half), bqB1 = ldB(qrowB, 32, half);
    const int qlaneA = qA * 16 + col, qlaneB = qB * 16 + col;

    const float* Mb = Mbuf + b * T_LEN;
    const float* Db = Dbuf + b * T_LEN;
    const int chA = qA / 2 + 1, chB = qB / 2 + 1;    // 32-key chunks per member
    v8f accA[4] = {}, accB[4] = {};

    for (int j = wave; j < chA + chB; j += 4) {      // wave-uniform split
        const bool isA  = j < chA;
        const int cidx  = isA ? j : (j - chA);
        const int qt    = isA ? qA : qB;
        const int q     = isA ? qlaneA : qlaneB;
        const v16bf bq0 = isA ? bqA0 : bqB0;
        const v16bf bq1 = isA ? bqA1 : bqB1;

        float p[16];
#pragma unroll
        for (int tt = 0; tt < 2; ++tt) {
            const int ktt = 2 * cidx + tt;
            if (ktt > qt) {                          // past-diagonal edge tile
#pragma unroll
                for (int r = 0; r < 8; ++r) p[8 * tt + r] = 0.0f;
                continue;
            }
            const __bf16* krow = Kb + base + (size_t)(ktt * 16 + col) * NDIM;
            __builtin_prefetch(krow + 8 * 16 * NDIM, 0, 1);   // next chunk (stride 4)
            v8f st = {};
            st = wmma_bf16(ldA(krow, 0, half), bq0, st);      // S^T = K . Q^T
            st = wmma_bf16(ldA(krow, 32, half), bq1, st);

            const float* mrow = Mb + ktt * 16 + 8 * half;
            const float* drow = Db + ktt * 16 + 8 * half;
            const v4f m0 = *(const v4f*)mrow, m1 = *(const v4f*)(mrow + 4);
            const v4f d0 = *(const v4f*)drow, d1 = *(const v4f*)(drow + 4);
#pragma unroll
            for (int r = 0; r < 8; ++r) {
                const int kg = ktt * 16 + r + 8 * half;       // key = M of S^T
                const float mm = (r < 4) ? m0[r] : m1[r - 4];
                const float dd = (r < 4) ? d0[r] : d1[r - 4];
                p[8 * tt + r] = (q >= kg) ? __expf(st[r] - mm) / dd : 0.0f;
            }
        }
        // Pack P into the 16-bit A-operand (identity mapping per lane)
        v16bf pa;
#pragma unroll
        for (int i = 0; i < 16; ++i) pa[i] = (__bf16)p[i];

        // B operands from transposed V: tokens [8h..8h+7]+[16+8h..23+8h] per lane
        const size_t vtok = (size_t)cidx * 32 + half * 8;
        const __bf16* v0 = Vt + ((size_t)(b * NDIM +  0 + col)) * T_LEN + vtok;
        const __bf16* v1 = Vt + ((size_t)(b * NDIM + 16 + col)) * T_LEN + vtok;
        const __bf16* v2 = Vt + ((size_t)(b * NDIM + 32 + col)) * T_LEN + vtok;
        const __bf16* v3 = Vt + ((size_t)(b * NDIM + 48 + col)) * T_LEN + vtok;
        const v16bf bv0 = cat16(ld8bf(v0), ld8bf(v0 + 16));
        const v16bf bv1 = cat16(ld8bf(v1), ld8bf(v1 + 16));
        const v16bf bv2 = cat16(ld8bf(v2), ld8bf(v2 + 16));
        const v16bf bv3 = cat16(ld8bf(v3), ld8bf(v3 + 16));
        if (isA) {
            accA[0] = wmma_bf16(pa, bv0, accA[0]);
            accA[1] = wmma_bf16(pa, bv1, accA[1]);
            accA[2] = wmma_bf16(pa, bv2, accA[2]);
            accA[3] = wmma_bf16(pa, bv3, accA[3]);
        } else {
            accB[0] = wmma_bf16(pa, bv0, accB[0]);
            accB[1] = wmma_bf16(pa, bv1, accB[1]);
            accB[2] = wmma_bf16(pa, bv2, accB[2]);
            accB[3] = wmma_bf16(pa, bv3, accB[3]);
        }
    }

    // Merge the 4 waves' partial PV sums in LDS (ds_add_f32)
#pragma unroll
    for (int nt = 0; nt < 4; ++nt)
#pragma unroll
        for (int r = 0; r < 8; ++r) {
            atomicAdd(&sOut[0][nt][r][lane], accA[nt][r]);
            atomicAdd(&sOut[1][nt][r][lane], accB[nt][r]);
        }
    __syncthreads();

    // Cooperative store: decode [pair][nt][r][lane] -> (q, d)
    for (int i = threadIdx.x; i < 2048; i += 128) {
        const int pp = i >> 10, rem = i & 1023;
        const int nt = rem >> 8, r = (rem >> 5) & 7, l = rem & 31;
        const int qt = pp ? qB : qA;
        const int qg = qt * 16 + r + 8 * (l >> 4);
        const int d  = nt * 16 + (l & 15);
        Out[((size_t)b * T_LEN + qg) * NDIM + d] = sflat[i];
    }
}

// ---------------------------------------------------------------------------
extern "C" void kernel_launch(void* const* d_in, const int* in_sizes, int n_in,
                              void* d_out, int out_size, void* d_ws, size_t ws_size,
                              hipStream_t stream) {
    (void)in_sizes; (void)n_in; (void)out_size; (void)ws_size;
    const float* Z  = (const float*)d_in[0];
    const float* Wq = (const float*)d_in[1];
    const float* Wk = (const float*)d_in[2];
    const float* Wv = (const float*)d_in[3];
    float* Out = (float*)d_out;

    char* ws = (char*)d_ws;
    __bf16* Qb = (__bf16*)(ws + 0);
    __bf16* Kb = (__bf16*)(ws + (2u << 20));
    __bf16* Vt = (__bf16*)(ws + (4u << 20));
    __bf16* Wt = (__bf16*)(ws + (6u << 20));
    float* Mbuf = (float*)(ws + (6u << 20) + 192 * 768 * 2);
    float* Dbuf = Mbuf + 8 * T_LEN;

    wtrans_kernel<<<(192 * C_LEN + 255) / 256, 256, 0, stream>>>(Wq, Wk, Wv, Wt);
    qkv_kernel<<<dim3(1024, 3), 128, 0, stream>>>(Z, Wt, Qb, Kb, Vt);
    colstats_kernel<<<dim3(64, 8), 128, 0, stream>>>(Qb, Kb, Mbuf, Dbuf);
    attn_out_kernel<<<dim3(64, 8), 128, 0, stream>>>(Qb, Kb, Vt, Mbuf, Dbuf, Out);
}
